// Transformer_DX_DG_DR_77747497992603
// MI455X (gfx1250) — compile-verified, matrix operand order checked by reading, not run on hardware
//
#include <hip/hip_runtime.h>
#include <hip/hip_bf16.h>
#include <stdint.h>

typedef _Float16 f16;
typedef __attribute__((ext_vector_type(16))) _Float16 v16h;
typedef __attribute__((ext_vector_type(8)))  _Float16 v8h;
typedef __attribute__((ext_vector_type(8)))  float    v8f;

// ---------------------------------------------------------------------------
// CDNA5 Tensor Data Mover availability (probe-confirmed builtin on ROCm 7.2)
// ---------------------------------------------------------------------------
#if __has_builtin(__builtin_amdgcn_tensor_load_to_lds) && \
    __has_builtin(__builtin_amdgcn_s_wait_tensorcnt)
#define USE_TDM 1
#else
#define USE_TDM 0
#endif

#if USE_TDM
typedef __attribute__((ext_vector_type(4))) unsigned int u32x4;
typedef __attribute__((ext_vector_type(4))) int i32x4;
typedef __attribute__((ext_vector_type(8))) int i32x8;

// Issue a TDM copy of a 2D f16 panel (tile_dim0 = K contiguous elems x 64
// rows, row stride = K elements -> fully contiguous) from global into LDS.
// D# layout per CDNA5 ISA ch.8: group0 = {count/lds/global/type},
// group1 = {data_size, tensor dims, tile dims, stride0}.
__device__ __forceinline__ void tdm_issue_bpanel(const f16* gsrc, int K,
                                                 unsigned lds_addr) {
  unsigned long long ga = (unsigned long long)(uintptr_t)gsrc;
  u32x4 g0;
  g0.x = 1u;                                               // count=1 (valid D#)
  g0.y = lds_addr;                                         // LDS byte address
  g0.z = (unsigned)ga;                                     // global_addr[31:0]
  g0.w = ((unsigned)(ga >> 32) & 0x1FFFFFFu) | (2u << 30); // addr[56:32]|type=2
  i32x8 g1;
  g1[0] = 0x00010000;              // workgroup_mask=0, data_size=1 (2 bytes)
  g1[1] = 0;                       // no atomic barrier; tensor_dim0 lo16 = 0
  g1[2] = 0x10;                    // tensor_dim0 = 1<<20 (hi16); dim1 lo16 = 0
  g1[3] = 0x10 | (K << 16);        // tensor_dim1 hi16; tile_dim0 = K elems
  g1[4] = 64;                      // tile_dim1 = 64 rows; tile_dim2 = 0
  g1[5] = K;                       // tensor_dim0_stride = K elems (lo32)
  g1[6] = 0;
  g1[7] = 0;
  i32x4 gz = {0, 0, 0, 0};
#if __clang_major__ >= 23
  i32x8 gz8 = {0, 0, 0, 0, 0, 0, 0, 0};
  __builtin_amdgcn_tensor_load_to_lds(g0, g1, gz, gz, gz8, 0);
#else
  __builtin_amdgcn_tensor_load_to_lds(g0, g1, gz, gz, 0);
#endif
}
#endif  // USE_TDM

// ---------------------------------------------------------------------------
// Helpers
// ---------------------------------------------------------------------------
__device__ __forceinline__ float wave_reduce_sum(float v) {
  #pragma unroll
  for (int off = 16; off > 0; off >>= 1) v += __shfl_xor(v, off, 32);
  return v;
}

__device__ __forceinline__ void atomicMaxF(float* addr, float val) {
  unsigned int* ua = (unsigned int*)addr;
  unsigned int old = *ua;
  while (__uint_as_float(old) < val) {
    unsigned int assumed = old;
    old = atomicCAS(ua, assumed, __float_as_uint(val));
    if (old == assumed) break;
  }
}

// ---------------------------------------------------------------------------
// Conversion / layout kernels
// ---------------------------------------------------------------------------
__global__ __launch_bounds__(256) void fill_f32_kernel(float* p, float v, long n) {
  long i = (long)blockIdx.x * 256 + threadIdx.x;
  if (i < n) p[i] = v;
}

__global__ __launch_bounds__(256) void cvt_f32_f16_kernel(const float* __restrict__ in,
                                                          f16* __restrict__ out, long n) {
  long i = (long)blockIdx.x * 256 + threadIdx.x;
  if (i < n) out[i] = (f16)in[i];
}

// W [K x Nc] (row major, f32) -> Wt [Nc x Kp] (f16, zero padded K..Kp)
__global__ __launch_bounds__(256) void transpose_cvt_kernel(const float* __restrict__ W,
                                                            f16* __restrict__ Wt,
                                                            int K, int Nc, int Kp) {
  long i = (long)blockIdx.x * 256 + threadIdx.x;
  if (i >= (long)Nc * Kp) return;
  int n = (int)(i / Kp), k = (int)(i % Kp);
  Wt[i] = (k < K) ? (f16)W[(size_t)k * Nc + n] : (f16)0.f;
}

// A [M x K] f32 -> Ap [M x Kp] f16 (zero padded)
__global__ __launch_bounds__(256) void pad_cvt_kernel(const float* __restrict__ in,
                                                      f16* __restrict__ out,
                                                      int M, int K, int Kp) {
  long i = (long)blockIdx.x * 256 + threadIdx.x;
  if (i >= (long)M * Kp) return;
  int m = (int)(i / Kp), k = (int)(i % Kp);
  out[i] = (k < K) ? (f16)in[(size_t)m * K + k] : (f16)0.f;
}

// ---------------------------------------------------------------------------
// WMMA GEMM:  C[M,N] = A[M,K] (f16, row major) * Bt[N,K]^T (f16) + bias
// One wave -> 16(M) x 64(N) tile; 8 waves/block -> 128 rows x 64 cols/block.
// The block's whole 64 x K B panel (<= 32 KB of the WGP's 320 KB LDS) is
// staged by ONE Tensor Data Mover transfer in the prologue (one
// s_wait_tensorcnt + one barrier), then the k-loop is barrier-free:
// batched ds_load_b128 of 4 B fragments -> 4 back-to-back v_wmma, with the
// global A fragment for k+32 prefetched one step ahead.
// K multiple of 32 and <= 256, N multiple of 64, M multiple of 16.
// ---------------------------------------------------------------------------
__global__ __launch_bounds__(256) void gemm_f16_wmma_kernel(
    const f16* __restrict__ A, const f16* __restrict__ Bt,
    const float* __restrict__ bias, float* __restrict__ C,
    int M, int N, int K) {
  const int lane = threadIdx.x & 31;
  const int wave = threadIdx.x >> 5;
  const int row0 = (blockIdx.x * 8 + wave) * 16;
  const int col0 = blockIdx.y * 64;
  const bool valid = row0 < M;
  const int rowc = valid ? row0 : (M - 16);  // clamp; wave 0 is always valid
  const int rr = lane & 15;   // row within A tile / col within B tile
  const int kh = lane >> 4;   // K-half selector per ISA 16-bit layout

#if USE_TDM
  __shared__ __align__(16) f16 bsm[64 * 256];  // full B panel, K <= 256

  if (wave == 0) {
    tdm_issue_bpanel(Bt + (size_t)col0 * K, K, (unsigned)(uintptr_t)&bsm[0]);
    __builtin_amdgcn_s_wait_tensorcnt(0);
  }
  __syncthreads();
#endif

  v8f acc[4] = {v8f{}, v8f{}, v8f{}, v8f{}};
  const f16* arow = A + (size_t)(rowc + rr) * K;

  // prologue: A fragment for k0 = 0
  v8h alo = *(const v8h*)(arow + kh * 8);
  v8h ahi = *(const v8h*)(arow + 16 + kh * 8);

  for (int k0 = 0; k0 < K; k0 += 32) {
    const bool last = (k0 + 32 >= K);

    // prefetch next A fragment (hides global latency behind the WMMAs)
    v8h nlo = {}, nhi = {};
    if (!last) {
      nlo = *(const v8h*)(arow + k0 + 32 + kh * 8);
      nhi = *(const v8h*)(arow + k0 + 48 + kh * 8);
    }

    v16h a;
    #pragma unroll
    for (int i = 0; i < 8; ++i) { a[i] = alo[i]; a[i + 8] = ahi[i]; }

    // batch all B fragment loads first -> single DS wait, then 4 WMMAs
    v8h blo[4], bhi[4];
    #pragma unroll
    for (int j = 0; j < 4; ++j) {
#if USE_TDM
      const f16* bp = &bsm[(size_t)(j * 16 + rr) * K + k0 + kh * 8];
#else
      const f16* bp = Bt + (size_t)(col0 + j * 16 + rr) * K + k0 + kh * 8;
#endif
      blo[j] = *(const v8h*)(bp);
      bhi[j] = *(const v8h*)(bp + 16);
    }
    #pragma unroll
    for (int j = 0; j < 4; ++j) {
      v16h b;
      #pragma unroll
      for (int i = 0; i < 8; ++i) { b[i] = blo[j][i]; b[i + 8] = bhi[j][i]; }
      acc[j] = __builtin_amdgcn_wmma_f32_16x16x32_f16(
          /*neg_a=*/false, a, /*neg_b=*/false, b,
          /*c_mod=*/(short)0, acc[j], /*reuse_a=*/false, /*reuse_b=*/false);
    }

    alo = nlo; ahi = nhi;
  }

  if (!valid) return;
  #pragma unroll
  for (int j = 0; j < 4; ++j) {
    int col = col0 + j * 16 + rr;
    float bb = bias ? bias[col] : 0.f;
    #pragma unroll
    for (int t = 0; t < 8; ++t) {
      int row = row0 + t + kh * 8;  // C layout: VGPR t -> M=t (+8 upper lanes)
      C[(size_t)row * N + col] = acc[j][t] + bb;
    }
  }
}

// ---------------------------------------------------------------------------
// Per-edge attention kernels  (H=4 heads, C=256 per head, HC=1024)
// ---------------------------------------------------------------------------
__global__ __launch_bounds__(256) void edge_logits_kernel(
    const float* __restrict__ q, const float* __restrict__ k,
    const float* __restrict__ ep, const int* __restrict__ ei, int E,
    float* __restrict__ logit, float* __restrict__ nmax) {
  int gw = (int)((blockIdx.x * 256 + threadIdx.x) >> 5);
  int lane = threadIdx.x & 31;
  if (gw >= E * 4) return;
  int e = gw >> 2, h = gw & 3;
  int src = ei[e], dst = ei[E + e];
  const float* qp = q + (size_t)dst * 1024 + h * 256;
  const float* kp = k + (size_t)src * 1024 + h * 256;
  const float* pp = ep + (size_t)e * 1024 + h * 256;
  float s = 0.f;
  #pragma unroll
  for (int c = lane; c < 256; c += 32) s += qp[c] * (kp[c] + pp[c]);
  s = wave_reduce_sum(s);
  if (lane == 0) {
    s *= 0.0625f;  // 1/sqrt(256)
    logit[(size_t)e * 4 + h] = s;
    atomicMaxF(&nmax[(size_t)dst * 4 + h], s);
  }
}

__global__ __launch_bounds__(256) void edge_exp_kernel(
    const float* __restrict__ logit, const int* __restrict__ ei, int E,
    const float* __restrict__ nmax, float* __restrict__ aexp,
    float* __restrict__ nsum) {
  int i = blockIdx.x * 256 + threadIdx.x;
  if (i >= E * 4) return;
  int e = i >> 2, h = i & 3;
  int dst = ei[E + e];
  float v = __expf(logit[i] - nmax[(size_t)dst * 4 + h]);
  aexp[i] = v;
  atomicAdd(&nsum[(size_t)dst * 4 + h], v);
}

__global__ __launch_bounds__(256) void edge_scatter_kernel(
    const float* __restrict__ v, const float* __restrict__ ep,
    const float* __restrict__ aexp, const float* __restrict__ nsum,
    const int* __restrict__ ei, int E, float* __restrict__ accum) {
  int e = blockIdx.x;
  int tid = threadIdx.x;
  int src = ei[e], dst = ei[E + e];
  #pragma unroll
  for (int t = 0; t < 4; ++t) {
    int ch = t * 256 + tid;  // head = t
    float a = aexp[(size_t)e * 4 + t] / nsum[(size_t)dst * 4 + t];
    float val = (v[(size_t)src * 1024 + ch] + ep[(size_t)e * 1024 + ch]) * a;
    atomicAdd(&accum[(size_t)dst * 1024 + ch], val);
  }
}

// mean over heads + root skip + ReLU; optionally emit f16 copy for next layer
__global__ __launch_bounds__(256) void combine_kernel(
    const float* __restrict__ accum, const float* __restrict__ root,
    float* __restrict__ hout, f16* __restrict__ hout16, int Nn) {
  long i = (long)blockIdx.x * 256 + threadIdx.x;
  if (i >= (long)Nn * 256) return;
  int n = (int)(i >> 8), c = (int)(i & 255);
  const float* a = accum + (size_t)n * 1024;
  float s = 0.25f * (a[c] + a[256 + c] + a[512 + c] + a[768 + c]) + root[i];
  s = fmaxf(s, 0.f);
  hout[i] = s;
  if (hout16) hout16[i] = (f16)s;
}

// ---------------------------------------------------------------------------
// Attentional aggregation + final FC
// ---------------------------------------------------------------------------
__global__ __launch_bounds__(256) void gate_dot_kernel(
    const float* __restrict__ h, const float* __restrict__ Wg,
    const float* __restrict__ bg, const int* __restrict__ bat, int Nn,
    float* __restrict__ glog, float* __restrict__ bmax) {
  int gw = (int)((blockIdx.x * 256 + threadIdx.x) >> 5);
  int lane = threadIdx.x & 31;
  if (gw >= Nn) return;
  float s = 0.f;
  #pragma unroll
  for (int c = lane; c < 256; c += 32) s += h[(size_t)gw * 256 + c] * Wg[c];
  s = wave_reduce_sum(s);
  if (lane == 0) {
    s += bg[0];
    glog[gw] = s;
    atomicMaxF(&bmax[bat[gw]], s);
  }
}

__global__ __launch_bounds__(256) void gate_exp_kernel(
    const float* __restrict__ glog, const int* __restrict__ bat, int Nn,
    const float* __restrict__ bmax, float* __restrict__ gexp,
    float* __restrict__ bsum) {
  int n = blockIdx.x * 256 + threadIdx.x;
  if (n >= Nn) return;
  int b = bat[n];
  float v = __expf(glog[n] - bmax[b]);
  gexp[n] = v;
  atomicAdd(&bsum[b], v);
}

__global__ __launch_bounds__(256) void pool_kernel(
    const float* __restrict__ h, const float* __restrict__ gexp,
    const float* __restrict__ bsum, const int* __restrict__ bat, int Nn,
    float* __restrict__ pooled) {
  long i = (long)blockIdx.x * 256 + threadIdx.x;
  if (i >= (long)Nn * 256) return;
  int n = (int)(i >> 8), c = (int)(i & 255);
  int b = bat[n];
  atomicAdd(&pooled[(size_t)b * 256 + c], h[i] * gexp[n] / bsum[b]);
}

__global__ __launch_bounds__(256) void final_kernel(
    const float* __restrict__ pooled, const float* __restrict__ g,
    const float* __restrict__ r, const float* __restrict__ Wfc,
    const float* __restrict__ bfc, float* __restrict__ out, int B) {
  int gw = (int)((blockIdx.x * 256 + threadIdx.x) >> 5);
  int lane = threadIdx.x & 31;
  if (gw >= B) return;
  float s = 0.f;
  #pragma unroll
  for (int c = lane; c < 512; c += 32) {
    float f = (c < 256) ? pooled[(size_t)gw * 256 + c]
            : (c < 384) ? g[(size_t)gw * 128 + (c - 256)]
                        : r[(size_t)gw * 128 + (c - 384)];
    s += f * Wfc[c];
  }
  s = wave_reduce_sum(s);
  if (lane == 0) out[gw] = s + bfc[0];
}

// ---------------------------------------------------------------------------
// Host orchestration
// ---------------------------------------------------------------------------
static inline unsigned nblk(long n) { return (unsigned)((n + 255) / 256); }

static void run_gemm(const f16* A, const f16* Bt, const float* bias, float* C,
                     int M, int Nc, int K, hipStream_t s) {
  dim3 grid((unsigned)((M + 127) / 128), (unsigned)(Nc / 64));
  gemm_f16_wmma_kernel<<<grid, 256, 0, s>>>(A, Bt, bias, C, M, Nc, K);
}

extern "C" void kernel_launch(void* const* d_in, const int* in_sizes, int n_in,
                              void* d_out, int out_size, void* d_ws, size_t ws_size,
                              hipStream_t stream) {
  (void)in_sizes; (void)n_in; (void)out_size; (void)ws_size;
  const int N = 40000, E = 80000, B = 1024;
  const int XD = 64, ED = 16, EDP = 32, HID = 256, HC = 1024, NH = 4;

  const float* x   = (const float*)d_in[0];
  const int*   ei  = (const int*)d_in[1];
  const float* ea  = (const float*)d_in[2];
  const int*   bat = (const int*)d_in[3];
  const float* g   = (const float*)d_in[4];
  const float* r   = (const float*)d_in[5];
  const float* Wq[2] = {(const float*)d_in[6],  (const float*)d_in[15]};
  const float* bq[2] = {(const float*)d_in[7],  (const float*)d_in[16]};
  const float* Wk[2] = {(const float*)d_in[8],  (const float*)d_in[17]};
  const float* bk[2] = {(const float*)d_in[9],  (const float*)d_in[18]};
  const float* Wv[2] = {(const float*)d_in[10], (const float*)d_in[19]};
  const float* bv[2] = {(const float*)d_in[11], (const float*)d_in[20]};
  const float* We[2] = {(const float*)d_in[12], (const float*)d_in[21]};
  const float* Ws[2] = {(const float*)d_in[13], (const float*)d_in[22]};
  const float* bs[2] = {(const float*)d_in[14], (const float*)d_in[23]};
  const float* Wg  = (const float*)d_in[24];
  const float* bg  = (const float*)d_in[25];
  const float* Wfc = (const float*)d_in[26];
  const float* bfc = (const float*)d_in[27];

  char* base = (char*)d_ws;
  size_t off = 0;
  auto alloc = [&](size_t bytes) -> void* {
    void* p = base + off;
    off += (bytes + 255) & ~(size_t)255;
    return p;
  };

  float* q      = (float*)alloc((size_t)N * HC * 4);
  float* kk     = (float*)alloc((size_t)N * HC * 4);
  float* vv     = (float*)alloc((size_t)N * HC * 4);
  float* epj    = (float*)alloc((size_t)E * HC * 4);
  float* accum  = (float*)alloc((size_t)N * HC * 4);
  float* root   = (float*)alloc((size_t)N * HID * 4);
  float* h1     = (float*)alloc((size_t)N * HID * 4);
  float* h2     = (float*)alloc((size_t)N * HID * 4);
  float* logit  = (float*)alloc((size_t)E * NH * 4);
  float* aexp   = (float*)alloc((size_t)E * NH * 4);
  float* nmax   = (float*)alloc((size_t)N * NH * 4);
  float* nsum   = (float*)alloc((size_t)N * NH * 4);
  float* glog   = (float*)alloc((size_t)N * 4);
  float* gexp   = (float*)alloc((size_t)N * 4);
  float* bmax   = (float*)alloc((size_t)B * 4);
  float* bsum   = (float*)alloc((size_t)B * 4);
  float* pooled = (float*)alloc((size_t)B * HID * 4);
  f16* x16  = (f16*)alloc((size_t)N * XD * 2);
  f16* h16  = (f16*)alloc((size_t)N * HID * 2);
  f16* ea16 = (f16*)alloc((size_t)E * EDP * 2);
  int KD[2] = {XD, HID};
  f16 *Wqt[2], *Wkt[2], *Wvt[2], *Wet[2], *Wst[2];
  for (int l = 0; l < 2; ++l) {
    Wqt[l] = (f16*)alloc((size_t)HC * KD[l] * 2);
    Wkt[l] = (f16*)alloc((size_t)HC * KD[l] * 2);
    Wvt[l] = (f16*)alloc((size_t)HC * KD[l] * 2);
    Wet[l] = (f16*)alloc((size_t)HC * EDP * 2);
    Wst[l] = (f16*)alloc((size_t)HID * KD[l] * 2);
  }

  // --- input conversions / weight transposes (f32 -> f16 [N x K] layout) ---
  cvt_f32_f16_kernel<<<nblk((long)N * XD), 256, 0, stream>>>(x, x16, (long)N * XD);
  pad_cvt_kernel<<<nblk((long)E * EDP), 256, 0, stream>>>(ea, ea16, E, ED, EDP);
  for (int l = 0; l < 2; ++l) {
    transpose_cvt_kernel<<<nblk((long)HC * KD[l]), 256, 0, stream>>>(Wq[l], Wqt[l], KD[l], HC, KD[l]);
    transpose_cvt_kernel<<<nblk((long)HC * KD[l]), 256, 0, stream>>>(Wk[l], Wkt[l], KD[l], HC, KD[l]);
    transpose_cvt_kernel<<<nblk((long)HC * KD[l]), 256, 0, stream>>>(Wv[l], Wvt[l], KD[l], HC, KD[l]);
    transpose_cvt_kernel<<<nblk((long)HC * EDP), 256, 0, stream>>>(We[l], Wet[l], ED, HC, EDP);
    transpose_cvt_kernel<<<nblk((long)HID * KD[l]), 256, 0, stream>>>(Ws[l], Wst[l], KD[l], HID, KD[l]);
  }

  // --- two TransformerConv layers ---
  for (int l = 0; l < 2; ++l) {
    const f16* in16 = (l == 0) ? x16 : h16;
    int K = KD[l];
    run_gemm(in16, Wqt[l], bq[l], q,  N, HC, K, stream);
    run_gemm(in16, Wkt[l], bk[l], kk, N, HC, K, stream);
    run_gemm(in16, Wvt[l], bv[l], vv, N, HC, K, stream);
    run_gemm(ea16, Wet[l], nullptr, epj, E, HC, EDP, stream);
    run_gemm(in16, Wst[l], bs[l], root, N, HID, K, stream);

    fill_f32_kernel<<<nblk((long)N * NH), 256, 0, stream>>>(nmax, -3.402823466e38f, (long)N * NH);
    fill_f32_kernel<<<nblk((long)N * NH), 256, 0, stream>>>(nsum, 0.f, (long)N * NH);
    fill_f32_kernel<<<nblk((long)N * HC), 256, 0, stream>>>(accum, 0.f, (long)N * HC);

    edge_logits_kernel<<<(unsigned)((long)E * NH * 32 / 256), 256, 0, stream>>>(q, kk, epj, ei, E, logit, nmax);
    edge_exp_kernel<<<nblk((long)E * NH), 256, 0, stream>>>(logit, ei, E, nmax, aexp, nsum);
    edge_scatter_kernel<<<E, 256, 0, stream>>>(vv, epj, aexp, nsum, ei, E, accum);
    combine_kernel<<<nblk((long)N * HID), 256, 0, stream>>>(
        accum, root, (l == 0) ? h1 : h2, (l == 0) ? h16 : (f16*)nullptr, N);
  }

  // --- attentional aggregation + final FC ---
  fill_f32_kernel<<<nblk(B), 256, 0, stream>>>(bmax, -3.402823466e38f, B);
  fill_f32_kernel<<<nblk(B), 256, 0, stream>>>(bsum, 0.f, B);
  fill_f32_kernel<<<nblk((long)B * HID), 256, 0, stream>>>(pooled, 0.f, (long)B * HID);

  gate_dot_kernel<<<(unsigned)((long)N * 32 / 256), 256, 0, stream>>>(h2, Wg, bg, bat, N, glog, bmax);
  gate_exp_kernel<<<nblk(N), 256, 0, stream>>>(glog, bat, N, bmax, gexp, bsum);
  pool_kernel<<<nblk((long)N * HID), 256, 0, stream>>>(h2, gexp, bsum, bat, N, pooled);
  final_kernel<<<(unsigned)((long)B * 32 / 256), 256, 0, stream>>>(pooled, g, r, Wfc, bfc, (float*)d_out, B);
}
